// ViT_GNN_Model_75634374082858
// MI455X (gfx1250) — compile-verified, matrix operand order checked by reading, not run on hardware
//
#include <hip/hip_runtime.h>
#include <hip/hip_bf16.h>

#define N_NODES 100864
#define N_EDGES 1613824
#define N_GRAPHS 512
#define IN_C 384
#define HID 128
#define OUT_C 2

typedef __attribute__((ext_vector_type(16))) __bf16 v16bf;
typedef __attribute__((ext_vector_type(8)))  float  v8f;

__device__ __forceinline__ void atomAddF(float* p, float v) {
    // lowers to global_atomic_add_f32 (no CAS loop)
    unsafeAtomicAdd(p, v);
}

// deg=1 (self-loop), pool=0, cnt=0
__global__ void k_init(float* __restrict__ deg, float* __restrict__ pool, float* __restrict__ cnt) {
    int i = blockIdx.x * blockDim.x + threadIdx.x;
    if (i < N_NODES) deg[i] = 1.0f;
    if (i < N_GRAPHS * HID) pool[i] = 0.0f;
    if (i < N_GRAPHS) cnt[i] = 0.0f;
}

__global__ void k_deg(const long long* __restrict__ dst, float* __restrict__ deg) {
    int e = blockIdx.x * blockDim.x + threadIdx.x;
    if (e < N_EDGES) atomAddF(&deg[(int)dst[e]], 1.0f);
}

__global__ void k_rsqrt(float* __restrict__ dis) {
    int i = blockIdx.x * blockDim.x + threadIdx.x;
    if (i < N_NODES) dis[i] = rsqrtf(dis[i]);
}

__global__ void k_zero4(float4* __restrict__ p, int n) {
    int i = blockIdx.x * blockDim.x + threadIdx.x;
    if (i < n) p[i] = make_float4(0.f, 0.f, 0.f, 0.f);
}

// G = dis ⊙ (X @ W), X:[N,K] f32, W:[K,128] f32 (converted to bf16 in LDS),
// 256 threads = 8 waves; wave -> 16 rows x 128 cols via 8 WMMA accumulators.
template <int KT>
__launch_bounds__(256)
__global__ void k_gemm_g(const float* __restrict__ X, const float* __restrict__ W,
                         const float* __restrict__ dis, float* __restrict__ Gout) {
    constexpr int K = KT * 32;
    __shared__ __bf16 wf[KT * 4096]; // [kt][nt][lane][16] fragment-ready

    const int tid = threadIdx.x;
    for (int idx = tid; idx < KT * 4096; idx += 256) {
        int j    = idx & 15;
        int lane = (idx >> 4) & 31;
        int nt   = (idx >> 9) & 7;
        int kt   = idx >> 12;
        int laneL = lane & 15;
        int half  = lane >> 4;
        int kl = half * 8 + ((j < 8) ? j : (16 + j - 8)); // ISA 16-bit frag K order
        int k  = kt * 32 + kl;
        int n  = nt * 16 + laneL;
        wf[idx] = (__bf16)W[k * HID + n];
    }
    __syncthreads();

    const int lane  = tid & 31;
    const int wave  = tid >> 5;
    const int laneL = lane & 15;
    const int half  = lane >> 4;
    const int rowBase = blockIdx.x * 128 + wave * 16;
    const int arow = rowBase + laneL;

    v8f acc[8] = {};

    for (int kt = 0; kt < KT; ++kt) {
        const float* xp = X + (size_t)arow * K + kt * 32 + half * 8;
        float4 f0 = *(const float4*)(xp + 0);
        float4 f1 = *(const float4*)(xp + 4);
        float4 f2 = *(const float4*)(xp + 16);
        float4 f3 = *(const float4*)(xp + 20);
        v16bf a;
        a[0]=(__bf16)f0.x; a[1]=(__bf16)f0.y; a[2]=(__bf16)f0.z;  a[3]=(__bf16)f0.w;
        a[4]=(__bf16)f1.x; a[5]=(__bf16)f1.y; a[6]=(__bf16)f1.z;  a[7]=(__bf16)f1.w;
        a[8]=(__bf16)f2.x; a[9]=(__bf16)f2.y; a[10]=(__bf16)f2.z; a[11]=(__bf16)f2.w;
        a[12]=(__bf16)f3.x; a[13]=(__bf16)f3.y; a[14]=(__bf16)f3.z; a[15]=(__bf16)f3.w;
#pragma unroll
        for (int nt = 0; nt < 8; ++nt) {
            v16bf b = *(const v16bf*)&wf[((kt * 8 + nt) * 32 + lane) * 16];
            acc[nt] = __builtin_amdgcn_wmma_f32_16x16x32_bf16(
                false, a, false, b, (short)0, acc[nt], false, false);
        }
    }

    float disv[8];
#pragma unroll
    for (int v = 0; v < 8; ++v) disv[v] = dis[rowBase + half * 8 + v];
#pragma unroll
    for (int nt = 0; nt < 8; ++nt) {
#pragma unroll
        for (int v = 0; v < 8; ++v) {
            // C/D layout: VGPR v -> M = v + 8*half ; N = laneL
            Gout[(size_t)(rowBase + half * 8 + v) * HID + nt * 16 + laneL] = disv[v] * acc[nt][v];
        }
    }
}

// S[dst] += G[src] ; one wave per edge, float4 per lane (4 f32 atomics)
__global__ void k_scatter(const long long* __restrict__ src, const long long* __restrict__ dst,
                          const float* __restrict__ Gm, float* __restrict__ S) {
    int e = blockIdx.x * 8 + (threadIdx.x >> 5);
    int c = (threadIdx.x & 31) * 4;
    long long s = src[e], d = dst[e];
    float4 g = *(const float4*)(Gm + (size_t)s * HID + c);
    float* ap = S + (size_t)d * HID + c;
    atomAddF(ap + 0, g.x); atomAddF(ap + 1, g.y);
    atomAddF(ap + 2, g.z); atomAddF(ap + 3, g.w);
}

// out = relu(dis[i]*(G+S) + b)   (out may alias G: same-index elementwise)
__global__ void k_relu_bias(const float* __restrict__ Gm, const float* __restrict__ S,
                            const float* __restrict__ dis, const float* __restrict__ b,
                            float* __restrict__ out) {
    int t = blockIdx.x * blockDim.x + threadIdx.x;
    if (t >= N_NODES * HID) return;
    int i = t >> 7, c = t & 127;
    float v = dis[i] * (Gm[t] + S[t]) + b[c];
    out[t] = fmaxf(v, 0.0f);
}

// final layer epilogue fused with mean-pool accumulation
__global__ void k_relu_pool(const float* __restrict__ Gm, const float* __restrict__ S,
                            const float* __restrict__ dis, const float* __restrict__ b,
                            const long long* __restrict__ batch,
                            float* __restrict__ pool, float* __restrict__ cnt) {
    int t = blockIdx.x * blockDim.x + threadIdx.x;
    if (t >= N_NODES * HID) return;
    int i = t >> 7, c = t & 127;
    float v = fmaxf(dis[i] * (Gm[t] + S[t]) + b[c], 0.0f);
    int g = (int)batch[i];
    atomAddF(&pool[g * HID + c], v);
    if (c == 0) atomAddF(&cnt[g], 1.0f);
}

__global__ void k_final(const float* __restrict__ pool, const float* __restrict__ cnt,
                        const float* __restrict__ Wlin, const float* __restrict__ blin,
                        float* __restrict__ out) {
    int t = blockIdx.x * blockDim.x + threadIdx.x;
    if (t >= N_GRAPHS * OUT_C) return;
    int g = t / OUT_C, o = t % OUT_C;
    float inv = 1.0f / fmaxf(cnt[g], 1.0f);
    float s = 0.f;
    for (int c = 0; c < HID; ++c) s += pool[g * HID + c] * Wlin[c * OUT_C + o];
    out[t] = s * inv + blin[o];
}

extern "C" void kernel_launch(void* const* d_in, const int* in_sizes, int n_in,
                              void* d_out, int out_size, void* d_ws, size_t ws_size,
                              hipStream_t stream) {
    const float*     x     = (const float*)d_in[0];
    const long long* ei    = (const long long*)d_in[1];
    const long long* batch = (const long long*)d_in[2];
    const float*     W1    = (const float*)d_in[3];
    const float*     b1    = (const float*)d_in[4];
    const float*     W2    = (const float*)d_in[5];
    const float*     b2    = (const float*)d_in[6];
    const float*     Wlin  = (const float*)d_in[7];
    const float*     blin  = (const float*)d_in[8];
    float*           out   = (float*)d_out;

    const long long* esrc = ei;
    const long long* edst = ei + N_EDGES;

    const size_t NH = (size_t)N_NODES * HID;            // 12,910,592 floats
    float* B0   = (float*)d_ws;                          // [N,128]
    float* B1   = B0 + NH;                               // [N,128]
    float* dis  = B1 + NH;                               // [N]
    float* pool = dis + N_NODES;                         // [G,128]
    float* cnt  = pool + (size_t)N_GRAPHS * HID;         // [G]
    // total ws use: ~104 MB

    const int T = 256;
    const int gNodes  = (N_NODES + T - 1) / T;
    const int gEdges  = (N_EDGES + T - 1) / T;
    const int gGemm   = N_NODES / 128;                   // 788, exact
    const int gScat   = N_EDGES / 8;                     // 201728, exact
    const int gElem   = (int)(NH / T);                   // 50432, exact
    const int gZero   = (int)(NH / 4 / T);               // 12608, exact

    // degree -> dis = rsqrt(deg)
    k_init  <<<gNodes, T, 0, stream>>>(dis, pool, cnt);
    k_deg   <<<gEdges, T, 0, stream>>>(edst, dis);
    k_rsqrt <<<gNodes, T, 0, stream>>>(dis);

    // ---- layer 1 ----
    k_gemm_g<12><<<gGemm, T, 0, stream>>>(x, W1, dis, B0);        // g1 = dis ⊙ (x@W1)
    k_zero4    <<<gZero, T, 0, stream>>>((float4*)B1, (int)(NH / 4));
    k_scatter  <<<gScat, T, 0, stream>>>(esrc, edst, B0, B1);     // B1 = Σ g1[src]
    k_relu_bias<<<gElem, T, 0, stream>>>(B0, B1, dis, b1, B0);    // B0 = relu1

    // ---- layer 2 ----
    k_gemm_g<4><<<gGemm, T, 0, stream>>>(B0, W2, dis, B1);        // g2 = dis ⊙ (relu1@W2)
    k_zero4   <<<gZero, T, 0, stream>>>((float4*)B0, (int)(NH / 4));
    k_scatter <<<gScat, T, 0, stream>>>(esrc, edst, B1, B0);      // B0 = Σ g2[src]
    k_relu_pool<<<gElem, T, 0, stream>>>(B1, B0, dis, b2, batch, pool, cnt);

    // ---- head ----
    k_final<<<(N_GRAPHS * OUT_C + T - 1) / T, T, 0, stream>>>(pool, cnt, Wlin, blin, out);
}
